// SMPLPoseEncoderSTGCN_25434796327526
// MI455X (gfx1250) — matrix operand order repaired and checked by reading, not compiled, not run on hardware
//
#include <hip/hip_runtime.h>
#include <hip/hip_bf16.h>

// ---------------------------------------------------------------------------
// SMPL ST-GCN pose encoder on gfx1250 (MI455X).
// Activations: f16, N-major [position][channel] with t-padded rows, so WMMA
// B-fragments are two contiguous 8-half runs (2x b128 loads / lane).
// Weights: pre-packed to f16 (K padded to 32, tcn taps split) so A-fragments
// are also contiguous (2x b128 loads / lane). All GEMMs: v_wmma_f32_16x16x32_f16,
// 4 O-tiles per wave sharing one B fragment (activation traffic / 4).
// ---------------------------------------------------------------------------

typedef __attribute__((ext_vector_type(16))) _Float16 v16h;
typedef __attribute__((ext_vector_type(8)))  _Float16 h8;
typedef __attribute__((ext_vector_type(8)))  float    v8f;

constexpr int  B_    = 32;
constexpr int  T_    = 256;
constexpr int  J_    = 24;
constexpr int  PT_   = T_ + 2;            // padded T (zero row before and after)
constexpr int  PLANE = PT_ * J_;          // 6192 (divisible by 16 and by J)
constexpr long NP    = (long)B_ * PLANE;  // 198144 padded positions
constexpr int  NT_PER_B = (T_ * J_) / 16; // 384 valid 16-wide tiles per image
constexpr int  NTILES   = B_ * NT_PER_B;  // 12288
constexpr int  NPOS     = B_ * T_ * J_;   // 196608 valid positions

// ---------------- fragment loaders (CDNA5 wave32 WMMA layouts) -------------
// A 16x32 f16: lane L -> M = L&15 ; element e -> K = kb + e + 8*(L>>4) + (e>=8?8:0)
// => per lane two contiguous 8-half runs at kb+8*half and kb+16+8*half.
__device__ __forceinline__ v16h load_frag_a(const _Float16* __restrict__ Wrow,
                                            int kb, int half) {
    const h8 lo = *(const h8*)(Wrow + kb + 8 * half);
    const h8 hi = *(const h8*)(Wrow + kb + 16 + 8 * half);
    v16h a;
#pragma unroll
    for (int e = 0; e < 8; ++e) { a[e] = lo[e]; a[e + 8] = hi[e]; }
    return a;
}

// B 32x16 f16: lane L -> N = L&15 ; element e -> K = kb + e + 8*(L>>4) + (e>=8?8:0)
// => per lane two contiguous 8-half runs at kb+8*half and kb+16+8*half.
__device__ __forceinline__ v16h load_frag_b(const _Float16* __restrict__ Xrow,
                                            int kb, int half) {
    const _Float16* p = Xrow + kb + 8 * half;
    const h8 lo = *(const h8*)(p);
    const h8 hi = *(const h8*)(p + 16);
    v16h b;
#pragma unroll
    for (int e = 0; e < 8; ++e) { b[e] = lo[e]; b[e + 8] = hi[e]; }
    return b;
}

// ---------------- weight packing (f32 -> f16, K-padded / tap-split) --------

__global__ void pack_w_kernel(const float* __restrict__ src, _Float16* __restrict__ dst,
                              int O, int Cin, int Cpad) {
    int id = blockIdx.x * blockDim.x + threadIdx.x;
    if (id >= O * Cpad) return;
    int c = id % Cpad, o = id / Cpad;
    dst[id] = (c < Cin) ? (_Float16)src[(size_t)o * Cin + c] : (_Float16)0.f;
}

// src [O][C][3] -> dst [3][O][C]
__global__ void pack_tcn_kernel(const float* __restrict__ src, _Float16* __restrict__ dst,
                                int O, int C) {
    int id = blockIdx.x * blockDim.x + threadIdx.x;
    if (id >= 3 * O * C) return;
    int c = id % C, r = id / C;
    int o = r % O, dt = r / O;
    dst[id] = (_Float16)src[((size_t)o * C + c) * 3 + dt];
}

// ---------------- adjacency prep (ADJ is a constant of the model) ----------

__global__ void adjprep_kernel(float* __restrict__ Aout, float* __restrict__ csum) {
    const int parent[J_] = {-1,0,0,0,1,2,3,4,5,6,7,8,9,9,9,12,13,14,16,17,18,19,20,21};
    __shared__ float A[J_][J_];
    int t = threadIdx.x;
    if (t < J_) {
        float deg = 0.f, row[J_];
        for (int k = 0; k < J_; ++k) {
            bool adj = (k == t) || (parent[k] == t) || (parent[t] == k);
            row[k] = adj ? 1.f : 0.f;
            deg += row[k];
        }
        deg = fmaxf(deg, 1.f);
        for (int k = 0; k < J_; ++k) A[t][k] = row[k] / deg;
    }
    __syncthreads();
    if (t < J_) {
        for (int k = 0; k < J_; ++k) Aout[t * J_ + k] = A[t][k];
        float cs = 0.f;
        for (int j = 0; j < J_; ++j) cs += A[j][t];
        csum[t] = cs;   // bias correction when ADJ is applied before the GEMM
    }
}

// ---------------- axis-angle -> rot6d, write [NP][32] f16 ------------------

__global__ void rot6d_kernel(const float* __restrict__ x, _Float16* __restrict__ F0) {
    long id = (long)blockIdx.x * blockDim.x + threadIdx.x;
    if (id >= NPOS) return;
    int j = (int)(id % J_);
    int t = (int)((id / J_) % T_);
    int b = (int)(id / ((long)T_ * J_));
    const float* p = x + (size_t)id * 3;
    float ax = p[0], ay = p[1], az = p[2];
    float ang = sqrtf(ax * ax + ay * ay + az * az);
    float inv = 1.f / (ang + 1e-8f);
    float ux = ax * inv, uy = ay * inv, uz = az * inv;
    float c = cosf(ang), s = sinf(ang), ic = 1.f - c;
    _Float16 row[32];
    row[0] = (_Float16)(c + ic * ux * ux);
    row[1] = (_Float16)(ic * ux * uy - s * uz);
    row[2] = (_Float16)(ic * uy * ux + s * uz);
    row[3] = (_Float16)(c + ic * uy * uy);
    row[4] = (_Float16)(ic * uz * ux - s * uy);
    row[5] = (_Float16)(ic * uz * uy + s * ux);
#pragma unroll
    for (int cc = 6; cc < 32; ++cc) row[cc] = (_Float16)0.f;
    size_t n = (size_t)b * PLANE + (size_t)(t + 1) * J_ + j;
    h8* dst = (h8*)(F0 + n * 32);
#pragma unroll
    for (int q = 0; q < 4; ++q) dst[q] = *(const h8*)(row + q * 8);
}

// ---------------- joint adjacency mix (valid rows only), N-major -----------

__global__ void adjmix_kernel(const _Float16* __restrict__ X, _Float16* __restrict__ Y,
                              const float* __restrict__ A, int C) {
    long id = (long)blockIdx.x * blockDim.x + threadIdx.x;
    long total = (long)C * B_ * T_ * J_;
    if (id >= total) return;
    int c = (int)(id % C);          // channel innermost -> coalesced
    long r1 = id / C;
    int k = (int)(r1 % J_);
    long r2 = r1 / J_;
    int t = (int)(r2 % T_);
    int b = (int)(r2 / T_);
    size_t base = (size_t)b * PLANE + (size_t)(t + 1) * J_;
    float s = 0.f;
#pragma unroll
    for (int j = 0; j < J_; ++j) s += (float)X[(base + j) * C + c] * A[j * J_ + k];
    Y[(base + k) * C + c] = (_Float16)s;
}

// ---------------- zero the t-pad rows of a GEMM output ---------------------

__global__ void padzero_kernel(_Float16* __restrict__ H, int C) {
    long id = (long)blockIdx.x * blockDim.x + threadIdx.x;
    long total = (long)B_ * 2 * J_ * C;
    if (id >= total) return;
    int c = (int)(id % C);
    long r1 = id / C;
    int j = (int)(r1 % J_);
    long r2 = r1 / J_;
    int side = (int)(r2 & 1);
    int b = (int)(r2 >> 1);
    size_t n = (size_t)b * PLANE + (side ? (size_t)(PT_ - 1) * J_ : 0) + j;
    H[n * C + c] = (_Float16)0.f;
}

// ---------------- GCN GEMM: Y = relu(bn1(W @ Xadj + b*csum[j])) ------------
// One wave computes 4 O-tiles (64 output channels) for one 16-wide N tile.

__global__ void __launch_bounds__(32)
gcn_gemm_kernel(const _Float16* __restrict__ W, const float* __restrict__ bias,
                const float* __restrict__ bn, const float* __restrict__ csum,
                const _Float16* __restrict__ X, _Float16* __restrict__ Y,
                int O, int Cin) {
    const int lane = threadIdx.x;
    const int half = lane >> 4;
    const int m    = lane & 15;
    const int b    = blockIdx.x / NT_PER_B;
    const int w    = blockIdx.x % NT_PER_B;
    const int o0   = blockIdx.y * 64;
    const long n0  = (long)b * PLANE + J_ + (long)w * 16;
    const long n   = n0 + m;
    const _Float16* Xrow = X + n * Cin;

    v8f acc[4] = {v8f{}, v8f{}, v8f{}, v8f{}};
    for (int kb = 0; kb < Cin; kb += 32) {
        v16h bf = load_frag_b(Xrow, kb, half);
#pragma unroll
        for (int ot = 0; ot < 4; ++ot) {
            v16h a = load_frag_a(W + (size_t)(o0 + ot * 16 + m) * Cin, kb, half);
            acc[ot] = __builtin_amdgcn_wmma_f32_16x16x32_f16(false, a, false, bf,
                                                             (short)0, acc[ot], false, false);
        }
    }
    const float cs = csum[(int)(n % J_)];   // PLANE % J_ == 0
#pragma unroll
    for (int ot = 0; ot < 4; ++ot) {
        int ob = o0 + ot * 16 + 8 * half;
        h8 outv;
#pragma unroll
        for (int r = 0; r < 8; ++r) {
            int o = ob + r;
            float g = bn[o], bb = bn[O + o], mm = bn[2 * O + o], v = bn[3 * O + o];
            float s = g * rsqrtf(v + 1e-5f);
            float val = (acc[ot][r] + bias[o] * cs) * s + (bb - mm * s);
            outv[r] = (_Float16)fmaxf(val, 0.f);
        }
        *(h8*)(Y + n * O + ob) = outv;
    }
}

// ---------------- residual projection GEMM: Y = W @ X + b ------------------

__global__ void __launch_bounds__(32)
res_gemm_kernel(const _Float16* __restrict__ W, const float* __restrict__ bias,
                const _Float16* __restrict__ X, _Float16* __restrict__ Y,
                int O, int Cin) {
    const int lane = threadIdx.x;
    const int half = lane >> 4;
    const int m    = lane & 15;
    const int b    = blockIdx.x / NT_PER_B;
    const int w    = blockIdx.x % NT_PER_B;
    const int o0   = blockIdx.y * 64;
    const long n   = (long)b * PLANE + J_ + (long)w * 16 + m;
    const _Float16* Xrow = X + n * Cin;

    v8f acc[4] = {v8f{}, v8f{}, v8f{}, v8f{}};
    for (int kb = 0; kb < Cin; kb += 32) {
        v16h bf = load_frag_b(Xrow, kb, half);
#pragma unroll
        for (int ot = 0; ot < 4; ++ot) {
            v16h a = load_frag_a(W + (size_t)(o0 + ot * 16 + m) * Cin, kb, half);
            acc[ot] = __builtin_amdgcn_wmma_f32_16x16x32_f16(false, a, false, bf,
                                                             (short)0, acc[ot], false, false);
        }
    }
#pragma unroll
    for (int ot = 0; ot < 4; ++ot) {
        int ob = o0 + ot * 16 + 8 * half;
        h8 outv;
#pragma unroll
        for (int r = 0; r < 8; ++r) outv[r] = (_Float16)(acc[ot][r] + bias[ob + r]);
        *(h8*)(Y + n * O + ob) = outv;
    }
}

// ------- temporal conv GEMM: Y = relu(bn2(conv3x1(H) + b) + R) -------------
// Wt packed tap-major: [3][O][Cin] f16.

__global__ void __launch_bounds__(32)
tcn_gemm_kernel(const _Float16* __restrict__ Wt, const float* __restrict__ bias,
                const float* __restrict__ bn, const _Float16* __restrict__ H,
                const _Float16* __restrict__ R, _Float16* __restrict__ Y,
                int O, int Cin) {
    const int lane = threadIdx.x;
    const int half = lane >> 4;
    const int m    = lane & 15;
    const int b    = blockIdx.x / NT_PER_B;
    const int w    = blockIdx.x % NT_PER_B;
    const int o0   = blockIdx.y * 64;
    const long n0  = (long)b * PLANE + J_ + (long)w * 16;
    const long n   = n0 + m;

    v8f acc[4] = {v8f{}, v8f{}, v8f{}, v8f{}};
#pragma unroll
    for (int dt = 0; dt < 3; ++dt) {
        const _Float16* Xrow = H + (n + (long)(dt - 1) * J_) * Cin;  // stays in padded plane
        const _Float16* Wtap = Wt + (size_t)dt * O * Cin;
        for (int kb = 0; kb < Cin; kb += 32) {
            v16h bf = load_frag_b(Xrow, kb, half);
#pragma unroll
            for (int ot = 0; ot < 4; ++ot) {
                v16h a = load_frag_a(Wtap + (size_t)(o0 + ot * 16 + m) * Cin, kb, half);
                acc[ot] = __builtin_amdgcn_wmma_f32_16x16x32_f16(false, a, false, bf,
                                                                 (short)0, acc[ot], false, false);
            }
        }
    }
#pragma unroll
    for (int ot = 0; ot < 4; ++ot) {
        int ob = o0 + ot * 16 + 8 * half;
        const h8 rv = *(const h8*)(R + n * O + ob);
        h8 outv;
#pragma unroll
        for (int r = 0; r < 8; ++r) {
            int o = ob + r;
            float g = bn[o], bb = bn[O + o], mm = bn[2 * O + o], v = bn[3 * O + o];
            float s = g * rsqrtf(v + 1e-5f);
            float val = (acc[ot][r] + bias[o]) * s + (bb - mm * s) + (float)rv[r];
            outv[r] = (_Float16)fmaxf(val, 0.f);
        }
        *(h8*)(Y + n * O + ob) = outv;
    }
}

// ---------------- mean-pool over (T,J) and linear head ---------------------

__global__ void pool_kernel(const _Float16* __restrict__ X, float* __restrict__ pooled) {
    int id = blockIdx.x * blockDim.x + threadIdx.x;
    if (id >= B_ * 256) return;
    int c = id % 256, b = id / 256;
    const _Float16* p = X + ((size_t)b * PLANE + J_) * 256 + c;
    float s = 0.f;
    for (int i = 0; i < T_ * J_; ++i) s += (float)p[(size_t)i * 256];
    pooled[id] = s * (1.f / (float)(T_ * J_));
}

__global__ void head_kernel(const float* __restrict__ pooled, const float* __restrict__ W,
                            const float* __restrict__ bias, float* __restrict__ out) {
    int id = blockIdx.x * blockDim.x + threadIdx.x;
    if (id >= B_ * 512) return;
    int o = id % 512, b = id / 512;
    const float* pw = W + (size_t)o * 256;
    const float* pp = pooled + (size_t)b * 256;
    float s = bias[o];
    for (int c = 0; c < 256; ++c) s += pp[c] * pw[c];
    out[id] = s;
}

// ---------------------------------------------------------------------------

extern "C" void kernel_launch(void* const* d_in, const int* in_sizes, int n_in,
                              void* d_out, int out_size, void* d_ws, size_t ws_size,
                              hipStream_t stream) {
    const float* x        = (const float*)d_in[0];
    const float* b1_gcn_w = (const float*)d_in[1];
    const float* b1_gcn_b = (const float*)d_in[2];
    const float* b1_bn1   = (const float*)d_in[3];
    const float* b1_tcn_w = (const float*)d_in[4];
    const float* b1_tcn_b = (const float*)d_in[5];
    const float* b1_bn2   = (const float*)d_in[6];
    const float* b1_res_w = (const float*)d_in[7];
    const float* b1_res_b = (const float*)d_in[8];
    const float* b2_gcn_w = (const float*)d_in[9];
    const float* b2_gcn_b = (const float*)d_in[10];
    const float* b2_bn1   = (const float*)d_in[11];
    const float* b2_tcn_w = (const float*)d_in[12];
    const float* b2_tcn_b = (const float*)d_in[13];
    const float* b2_bn2   = (const float*)d_in[14];
    const float* b3_gcn_w = (const float*)d_in[15];
    const float* b3_gcn_b = (const float*)d_in[16];
    const float* b3_bn1   = (const float*)d_in[17];
    const float* b3_tcn_w = (const float*)d_in[18];
    const float* b3_tcn_b = (const float*)d_in[19];
    const float* b3_bn2   = (const float*)d_in[20];
    const float* b3_res_w = (const float*)d_in[21];
    const float* b3_res_b = (const float*)d_in[22];
    const float* out_w    = (const float*)d_in[23];
    const float* out_b    = (const float*)d_in[24];

    char* ws = (char*)d_ws;
    size_t off = 0;
    auto take = [&](size_t bytes) { size_t o = off; off += (bytes + 255) & ~(size_t)255; return o; };
    float*    ADJn   = (float*)(ws + take(J_ * J_ * sizeof(float)));
    float*    CSUM   = (float*)(ws + take(J_ * sizeof(float)));
    float*    POOLED = (float*)(ws + take((size_t)B_ * 256 * sizeof(float)));
    // packed f16 weights
    _Float16* W1g = (_Float16*)(ws + take((size_t)128 * 32  * 2));
    _Float16* W1r = (_Float16*)(ws + take((size_t)128 * 32  * 2));
    _Float16* W2g = (_Float16*)(ws + take((size_t)128 * 128 * 2));
    _Float16* W3g = (_Float16*)(ws + take((size_t)256 * 128 * 2));
    _Float16* W3r = (_Float16*)(ws + take((size_t)256 * 128 * 2));
    _Float16* Wt1 = (_Float16*)(ws + take((size_t)3 * 128 * 128 * 2));
    _Float16* Wt2 = (_Float16*)(ws + take((size_t)3 * 128 * 128 * 2));
    _Float16* Wt3 = (_Float16*)(ws + take((size_t)3 * 256 * 256 * 2));
    // f16 activations, N-major
    _Float16* F0  = (_Float16*)(ws + take((size_t)NP * 32  * 2));
    _Float16* XA  = (_Float16*)(ws + take((size_t)NP * 128 * 2));
    _Float16* Hb  = (_Float16*)(ws + take((size_t)NP * 256 * 2));
    _Float16* Rb  = (_Float16*)(ws + take((size_t)NP * 256 * 2));
    _Float16* A1  = (_Float16*)(ws + take((size_t)NP * 256 * 2));
    _Float16* A2  = (_Float16*)(ws + take((size_t)NP * 128 * 2));
    (void)ws_size; (void)in_sizes; (void)n_in; (void)out_size;

    auto cdiv = [](long a, long b) { return (int)((a + b - 1) / b); };

    adjprep_kernel<<<1, 32, 0, stream>>>(ADJn, CSUM);
    pack_w_kernel<<<cdiv(128 * 32, 256), 256, 0, stream>>>(b1_gcn_w, W1g, 128, 6, 32);
    pack_w_kernel<<<cdiv(128 * 32, 256), 256, 0, stream>>>(b1_res_w, W1r, 128, 6, 32);
    pack_w_kernel<<<cdiv(128 * 128, 256), 256, 0, stream>>>(b2_gcn_w, W2g, 128, 128, 128);
    pack_w_kernel<<<cdiv(256 * 128, 256), 256, 0, stream>>>(b3_gcn_w, W3g, 256, 128, 128);
    pack_w_kernel<<<cdiv(256 * 128, 256), 256, 0, stream>>>(b3_res_w, W3r, 256, 128, 128);
    pack_tcn_kernel<<<cdiv(3 * 128 * 128, 256), 256, 0, stream>>>(b1_tcn_w, Wt1, 128, 128);
    pack_tcn_kernel<<<cdiv(3 * 128 * 128, 256), 256, 0, stream>>>(b2_tcn_w, Wt2, 128, 128);
    pack_tcn_kernel<<<cdiv(3 * 256 * 256, 256), 256, 0, stream>>>(b3_tcn_w, Wt3, 256, 256);

    rot6d_kernel<<<cdiv(NPOS, 256), 256, 0, stream>>>(x, F0);

    // ---- block 1: 6(->32 padded) -> 128 ----
    adjmix_kernel<<<cdiv((long)32 * NPOS, 256), 256, 0, stream>>>(F0, XA, ADJn, 32);
    gcn_gemm_kernel<<<dim3(NTILES, 128 / 64), 32, 0, stream>>>(
        W1g, b1_gcn_b, b1_bn1, CSUM, XA, Hb, 128, 32);
    padzero_kernel<<<cdiv((long)B_ * 2 * J_ * 128, 256), 256, 0, stream>>>(Hb, 128);
    res_gemm_kernel<<<dim3(NTILES, 128 / 64), 32, 0, stream>>>(W1r, b1_res_b, F0, Rb, 128, 32);
    tcn_gemm_kernel<<<dim3(NTILES, 128 / 64), 32, 0, stream>>>(
        Wt1, b1_tcn_b, b1_bn2, Hb, Rb, A1, 128, 128);

    // ---- block 2: 128 -> 128 (identity residual = A1) ----
    adjmix_kernel<<<cdiv((long)128 * NPOS, 256), 256, 0, stream>>>(A1, XA, ADJn, 128);
    gcn_gemm_kernel<<<dim3(NTILES, 128 / 64), 32, 0, stream>>>(
        W2g, b2_gcn_b, b2_bn1, CSUM, XA, Hb, 128, 128);
    padzero_kernel<<<cdiv((long)B_ * 2 * J_ * 128, 256), 256, 0, stream>>>(Hb, 128);
    tcn_gemm_kernel<<<dim3(NTILES, 128 / 64), 32, 0, stream>>>(
        Wt2, b2_tcn_b, b2_bn2, Hb, A1, A2, 128, 128);

    // ---- block 3: 128 -> 256 ----
    adjmix_kernel<<<cdiv((long)128 * NPOS, 256), 256, 0, stream>>>(A2, XA, ADJn, 128);
    gcn_gemm_kernel<<<dim3(NTILES, 256 / 64), 32, 0, stream>>>(
        W3g, b3_gcn_b, b3_bn1, CSUM, XA, Hb, 256, 128);
    padzero_kernel<<<cdiv((long)B_ * 2 * J_ * 256, 256), 256, 0, stream>>>(Hb, 256);
    res_gemm_kernel<<<dim3(NTILES, 256 / 64), 32, 0, stream>>>(W3r, b3_res_b, A2, Rb, 256, 128);
    tcn_gemm_kernel<<<dim3(NTILES, 256 / 64), 32, 0, stream>>>(
        Wt3, b3_tcn_b, b3_bn2, Hb, Rb, A1, 256, 256);

    // ---- pool + head ----
    pool_kernel<<<cdiv((long)B_ * 256, 256), 256, 0, stream>>>(A1, POOLED);
    head_kernel<<<cdiv((long)B_ * 512, 256), 256, 0, stream>>>(POOLED, out_w, out_b, (float*)d_out);
}
